// desakt_50629074485961
// MI455X (gfx1250) — matrix operand order repaired and checked by reading, not compiled, not run on hardware
//
#include <hip/hip_runtime.h>
#include <hip/hip_bf16.h>
#include <math.h>

#define SEQL 1024
#define BATCH 8
#define EMBD 512
#define NHEAD 8
#define HDIM 64
#define NUMC 10000
#define BL (BATCH*SEQL)
#define ROWE ((size_t)BL*EMBD)

typedef __attribute__((ext_vector_type(16))) __bf16 v16bf;
typedef __attribute__((ext_vector_type(8)))  __bf16 bf8;
typedef __attribute__((ext_vector_type(8)))  float  v8f;

#if defined(__gfx1250__) && __has_builtin(__builtin_amdgcn_global_load_async_to_lds_b128)
#define HAS_ASYNC_LDS 1
#else
#define HAS_ASYNC_LDS 0
#endif

#if HAS_ASYNC_LDS
#define GLOBAL_AS __attribute__((address_space(1)))
#define LDS_AS    __attribute__((address_space(3)))
typedef int v4i_async __attribute__((vector_size(16)));   // matches builtin's pointee type
__device__ __forceinline__ void async_cp_b128(const void* g, void* l) {
  __builtin_amdgcn_global_load_async_to_lds_b128(
      (GLOBAL_AS v4i_async*)g, (LDS_AS v4i_async*)l, 0, 0);
}
template <int N>
__device__ __forceinline__ void wait_async() {
#if __has_builtin(__builtin_amdgcn_s_wait_asynccnt)
  __builtin_amdgcn_s_wait_asynccnt(N);
#else
  if constexpr (N == 0) asm volatile("s_wait_asynccnt 0x0" ::: "memory");
  else                  asm volatile("s_wait_asynccnt 0x1" ::: "memory");
#endif
}
#endif

__device__ inline v8f wmma_bf16(v16bf a, v16bf b, v8f c) {
  // D = A(16x32 bf16) * B(32x16 bf16) + C(16x16 f32)
  return __builtin_amdgcn_wmma_f32_16x16x32_bf16(false, a, false, b, (short)0, c, false, false);
}
__device__ inline v16bf join8(bf8 lo, bf8 hi) {
  return __builtin_shufflevector(lo, hi, 0,1,2,3,4,5,6,7,8,9,10,11,12,13,14,15);
}
__device__ inline float rmax16(float v) {
#pragma unroll
  for (int o = 1; o < 16; o <<= 1) v = fmaxf(v, __shfl_xor(v, o, 32));
  return v;
}
__device__ inline float rsum16(float v) {
#pragma unroll
  for (int o = 1; o < 16; o <<= 1) v += __shfl_xor(v, o, 32);
  return v;
}
__device__ inline float rsum32(float v) {
#pragma unroll
  for (int o = 1; o < 32; o <<= 1) v += __shfl_xor(v, o, 32);
  return v;
}

// ---------------- small prep kernels ----------------

__global__ void cvt_bf16_kernel(const float* __restrict__ src, __bf16* __restrict__ dst, int n) {
  for (int i = blockIdx.x * blockDim.x + threadIdx.x; i < n; i += gridDim.x * blockDim.x)
    dst[i] = (__bf16)src[i];
}

// one block of 1024 threads: gumbel-softmax mask m[L] and mask_L1 = sum|alphas|
__global__ void mask_kernel(const float* __restrict__ alphas, const float* __restrict__ gumbel,
                            float* __restrict__ m_out, float* __restrict__ l1_out) {
  __shared__ float red[32];
  int t = threadIdx.x;
  float a0 = alphas[2 * t] + gumbel[2 * t];
  float a1 = alphas[2 * t + 1] + gumbel[2 * t + 1];
  float mx = fmaxf(a0, a1);
  float e0 = expf(a0 - mx), e1 = expf(a1 - mx);
  float soft0 = e0 / (e0 + e1);
  float hard0 = (a0 >= a1) ? 1.0f : 0.0f;          // argmax (ties -> first), one-hot col 0
  m_out[t] = (hard0 - soft0) + soft0;               // straight-through value
  float v = fabsf(alphas[2 * t]) + fabsf(alphas[2 * t + 1]);
  v = rsum32(v);
  if ((t & 31) == 0) red[t >> 5] = v;
  __syncthreads();
  if (t < 32) {
    float w = red[t];
    w = rsum32(w);
    if (t == 0) l1_out[0] = w;
  }
}

// x = inter_emb[q + C*r] + pos,  qsh = ex_emb[qry] ; fp32 + bf16 copies
__global__ void embed_kernel(const int* __restrict__ q, const int* __restrict__ r,
                             const int* __restrict__ qry,
                             const float* __restrict__ inter, const float* __restrict__ ex,
                             const float* __restrict__ pos,
                             float* __restrict__ xf, __bf16* __restrict__ xh,
                             float* __restrict__ qf, __bf16* __restrict__ qh) {
  size_t n = ROWE;
  for (size_t i = blockIdx.x * (size_t)blockDim.x + threadIdx.x; i < n;
       i += (size_t)gridDim.x * blockDim.x) {
    int row = (int)(i >> 9);
    int e = (int)(i & 511);
    int l = row & (SEQL - 1);
    float xv = inter[(size_t)(q[row] + NUMC * r[row]) * EMBD + e] + pos[(size_t)l * EMBD + e];
    float qv = ex[(size_t)qry[row] * EMBD + e];
    xf[i] = xv; xh[i] = (__bf16)xv;
    qf[i] = qv; qh[i] = (__bf16)qv;
  }
}

// ---------------- WMMA GEMM:  Y[M=8192, N] = A[M,512] * W[N,512]^T + bias ----------------
// grid = (M/128, N/64), block = 256 (8 waves, one 16x64 output tile per wave)
// W tile double-buffered in LDS via async global->LDS copies.
__global__ __launch_bounds__(256) void gemm_kernel(
    const __bf16* __restrict__ A, const __bf16* __restrict__ W,
    const float* __restrict__ bias,
    float* __restrict__ Yf, __bf16* __restrict__ Yh, int relu) {
  __shared__ __align__(16) __bf16 smB[2][64 * 40];   // W tile [n=64][k=32] + pad, x2 buffers
  const int t = threadIdx.x;
  const int wave = t >> 5;
  const int lane = t & 31;
  const int hw = lane >> 4;      // half-wave select
  const int ln = lane & 15;
  const int Mbase = blockIdx.x * 128 + wave * 16;
  const int Nbase = blockIdx.y * 64;

  v8f c[4] = {};
  const __bf16* Arow = A + (size_t)(Mbase + ln) * EMBD;

  // this thread's slice of the W tile: one b128 per tile
  const int bn = t >> 2;
  const int bks = (t & 3) * 8;
  const __bf16* wsrc = &W[(size_t)(Nbase + bn) * EMBD + bks];
  __bf16* bdst0 = &smB[0][bn * 40 + bks];
  __bf16* bdst1 = &smB[1][bn * 40 + bks];

#if HAS_ASYNC_LDS
  async_cp_b128(wsrc, bdst0);               // prefetch tile 0
#endif

  for (int k0 = 0; k0 < EMBD; k0 += 32) {
    const int cur = (k0 >> 5) & 1;
    __syncthreads();                        // everyone done reading buffer cur^1
#if HAS_ASYNC_LDS
    if (k0 + 32 < EMBD) {                   // issue next tile, then wait for current
      async_cp_b128(wsrc + k0 + 32, cur ? bdst0 : bdst1);
      wait_async<1>();
    } else {
      wait_async<0>();
    }
#else
    *(bf8*)&smB[cur][bn * 40 + bks] = *(const bf8*)&wsrc[k0];
#endif
    __syncthreads();                        // tile visible to all waves

    if (k0 + 64 < EMBD) __builtin_prefetch(&Arow[k0 + 64], 0, 1);
    // A fragment: lane row = ln; halves per gfx1250 16-bit A layout
    bf8 alo = *(const bf8*)&Arow[k0 + hw * 8];
    bf8 ahi = *(const bf8*)&Arow[k0 + 16 + hw * 8];
    v16bf a = join8(alo, ahi);
#pragma unroll
    for (int ns = 0; ns < 4; ns++) {
      int n = ns * 16 + ln;
      bf8 blo = *(const bf8*)&smB[cur][n * 40 + hw * 8];
      bf8 bhi = *(const bf8*)&smB[cur][n * 40 + 16 + hw * 8];
      c[ns] = wmma_bf16(a, join8(blo, bhi), c[ns]);
    }
  }

#pragma unroll
  for (int ns = 0; ns < 4; ns++) {
    int col = Nbase + ns * 16 + ln;
    float bv = bias[col];
#pragma unroll
    for (int rr = 0; rr < 8; rr++) {
      int orow = Mbase + hw * 8 + rr;           // C layout: rows = hw*8 + vgpr
      float f = c[ns][rr] + bv;
      if (relu) f = fmaxf(f, 0.0f);
      size_t o = (size_t)orow * EMBD + col;
      if (Yf) Yf[o] = f;
      if (Yh) Yh[o] = (__bf16)f;
    }
  }
}

// ---------------- WMMA flash attention ----------------
// grid = (B*H, L/64), block = 128 (4 waves, 16 query rows each)
__global__ __launch_bounds__(128) void attn_kernel(
    const __bf16* __restrict__ Q, const __bf16* __restrict__ K, const __bf16* __restrict__ V,
    const float* __restrict__ mvec, __bf16* __restrict__ O) {
  __shared__ __align__(16) float  smM[SEQL];
  __shared__ __align__(16) __bf16 smK[32 * 72];     // K tile [j=32][d=64]+pad
  __shared__ __align__(16) __bf16 smVt[64 * 40];    // V tile transposed [d=64][j=32]+pad
  __shared__ __align__(16) __bf16 smP[4 * 16 * 32]; // per-wave P bounce

  const int t = threadIdx.x;
  const int wave = t >> 5, lane = t & 31, hw = lane >> 4, ln = lane & 15;
  const int bh = blockIdx.x;
  const int b = bh >> 3, h = bh & 7;
  const int ibase = blockIdx.y * 64;
  const int iw = ibase + wave * 16;
  const size_t rowOff = (size_t)b * SEQL;
  const int hd0 = h * HDIM;
  const float scale = 0.125f;   // 1/sqrt(64)
  const float NEG = -3.0e38f;

  for (int s = t; s < SEQL; s += 128) smM[s] = mvec[s];
  __syncthreads();

  v8f o_acc[4] = {};
  float rm[8], rl[8];
#pragma unroll
  for (int i = 0; i < 8; i++) { rm[i] = NEG; rl[i] = 0.0f; }

  const int jmax = ibase + 64;            // causal: no j beyond block's last row
  __bf16* myP = &smP[wave * 16 * 32];

  for (int j0 = 0; j0 < jmax; j0 += 32) {
    __syncthreads();                      // previous iteration done with smK/smVt
    {   // stage K tile (async mem->LDS) and V tile (transposed via VGPRs)
      int jl = t >> 2;
      int ds = (t & 3) * 16;
      const __bf16* ksrc = &K[(rowOff + j0 + jl) * EMBD + hd0 + ds];
#if HAS_ASYNC_LDS
      async_cp_b128(ksrc,     &smK[jl * 72 + ds]);
      async_cp_b128(ksrc + 8, &smK[jl * 72 + ds + 8]);
#else
      *(bf8*)&smK[jl * 72 + ds]     = *(const bf8*)&ksrc[0];
      *(bf8*)&smK[jl * 72 + ds + 8] = *(const bf8*)&ksrc[8];
#endif
      const __bf16* vsrc = &V[(rowOff + j0 + jl) * EMBD + hd0 + ds];
      bf8 v0 = *(const bf8*)&vsrc[0];
      bf8 v1 = *(const bf8*)&vsrc[8];
#pragma unroll
      for (int ii = 0; ii < 8; ii++) {
        smVt[(ds + ii) * 40 + jl]     = v0[ii];
        smVt[(ds + 8 + ii) * 40 + jl] = v1[ii];
      }
    }
#if HAS_ASYNC_LDS
    wait_async<0>();
#endif
    __syncthreads();

    // S = Q * K^T for two 16-wide j sub-tiles, K-dim = 64 (2 WMMAs each)
    v8f s01[2];
#pragma unroll
    for (int sub = 0; sub < 2; sub++) {
      v8f cacc = {};
#pragma unroll
      for (int dk = 0; dk < 2; dk++) {
        int d0 = dk * 32;
        const __bf16* qrow = &Q[(rowOff + iw + ln) * EMBD + hd0 + d0];
        bf8 alo = *(const bf8*)&qrow[hw * 8];
        bf8 ahi = *(const bf8*)&qrow[16 + hw * 8];
        int jl = sub * 16 + ln;
        bf8 blo = *(const bf8*)&smK[jl * 72 + d0 + hw * 8];
        bf8 bhi = *(const bf8*)&smK[jl * 72 + d0 + 16 + hw * 8];
        cacc = wmma_bf16(join8(alo, ahi), join8(blo, bhi), cacc);
      }
      s01[sub] = cacc;
    }

    // online softmax per accumulator row
    float al[8];
#pragma unroll
    for (int rr = 0; rr < 8; rr++) {
      int ig = iw + hw * 8 + rr;
      float sv[2];
#pragma unroll
      for (int sub = 0; sub < 2; sub++) {
        int jg = j0 + sub * 16 + ln;
        int d = ig - jg;
        int dc = d < 0 ? 0 : d;
        bool ok = (d >= 0) && (smM[dc] != 0.0f);   // causal & distance-gated mask
        sv[sub] = ok ? s01[sub][rr] * scale : NEG;
      }
      float tmax = rmax16(fmaxf(sv[0], sv[1]));
      float mnew = fmaxf(rm[rr], tmax);
      float alpha = expf(rm[rr] - mnew);
      float p0 = expf(sv[0] - mnew);
      float p1 = expf(sv[1] - mnew);
      float rs = rsum16(p0 + p1);
      rl[rr] = rl[rr] * alpha + rs;
      rm[rr] = mnew;
      al[rr] = alpha;
      int mrow = hw * 8 + rr;
      myP[mrow * 32 + ln]      = (__bf16)p0;
      myP[mrow * 32 + 16 + ln] = (__bf16)p1;
    }
#pragma unroll
    for (int dsub = 0; dsub < 4; dsub++)
#pragma unroll
      for (int rr = 0; rr < 8; rr++)
        o_acc[dsub][rr] *= al[rr];

    asm volatile("s_wait_dscnt 0" ::: "memory");   // P stores -> P fragment loads (same wave)

    // O += P(16x32) * V(32x64), 4 d sub-tiles
    bf8 plo = *(const bf8*)&myP[ln * 32 + hw * 8];
    bf8 phi = *(const bf8*)&myP[ln * 32 + 16 + hw * 8];
    v16bf pA = join8(plo, phi);
#pragma unroll
    for (int dsub = 0; dsub < 4; dsub++) {
      int n = dsub * 16 + ln;
      bf8 vlo = *(const bf8*)&smVt[n * 40 + hw * 8];
      bf8 vhi = *(const bf8*)&smVt[n * 40 + 16 + hw * 8];
      o_acc[dsub] = wmma_bf16(pA, join8(vlo, vhi), o_acc[dsub]);
    }
  }

#pragma unroll
  for (int dsub = 0; dsub < 4; dsub++) {
#pragma unroll
    for (int rr = 0; rr < 8; rr++) {
      int orow = iw + hw * 8 + rr;
      float val = o_acc[dsub][rr] / rl[rr];
      O[(rowOff + orow) * EMBD + hd0 + dsub * 16 + ln] = (__bf16)val;
    }
  }
}

// ---------------- LayerNorm(a + b) * s + bb, wave per row ----------------
__global__ __launch_bounds__(256) void ln_kernel(
    const float* __restrict__ a, const float* __restrict__ bsrc,
    const float* __restrict__ s, const float* __restrict__ bb,
    float* __restrict__ outf, __bf16* __restrict__ outh) {
  int wave = threadIdx.x >> 5, lane = threadIdx.x & 31;
  int row = blockIdx.x * 8 + wave;
  const float* pa = a + (size_t)row * EMBD;
  const float* pb = bsrc + (size_t)row * EMBD;
  float v[16];
  float sum = 0.f;
#pragma unroll
  for (int i = 0; i < 16; i++) { int e = lane + 32 * i; v[i] = pa[e] + pb[e]; sum += v[i]; }
  sum = rsum32(sum);
  float mean = sum * (1.0f / EMBD);
  float var = 0.f;
#pragma unroll
  for (int i = 0; i < 16; i++) { float d = v[i] - mean; var += d * d; }
  var = rsum32(var) * (1.0f / EMBD);
  float inv = rsqrtf(var + 1e-5f);
#pragma unroll
  for (int i = 0; i < 16; i++) {
    int e = lane + 32 * i;
    float f = (v[i] - mean) * inv * s[e] + bb[e];
    outf[(size_t)row * EMBD + e] = f;
    outh[(size_t)row * EMBD + e] = (__bf16)f;
  }
}

// ---------------- p = sigmoid(x . pred_w + pred_b), wave per row ----------------
__global__ __launch_bounds__(256) void pred_kernel(
    const float* __restrict__ x, const float* __restrict__ pw, const float* __restrict__ pb,
    float* __restrict__ out) {
  int wave = threadIdx.x >> 5, lane = threadIdx.x & 31;
  int row = blockIdx.x * 8 + wave;
  const float* px = x + (size_t)row * EMBD;
  float v = 0.f;
#pragma unroll
  for (int i = 0; i < 16; i++) { int e = lane + 32 * i; v += px[e] * pw[e]; }
  v = rsum32(v);
  if (lane == 0) out[row] = 1.0f / (1.0f + expf(-(v + pb[0])));
}

// ---------------- host launcher ----------------

extern "C" void kernel_launch(void* const* d_in, const int* in_sizes, int n_in,
                              void* d_out, int out_size, void* d_ws, size_t ws_size,
                              hipStream_t stream) {
  (void)in_sizes; (void)n_in; (void)out_size; (void)ws_size;
  const int*   q      = (const int*)d_in[0];
  const int*   r      = (const int*)d_in[1];
  const int*   qry    = (const int*)d_in[2];
  const float* alphas = (const float*)d_in[3];
  const float* gumbel = (const float*)d_in[4];
  const float* inter  = (const float*)d_in[5];
  const float* ex     = (const float*)d_in[6];
  const float* pos    = (const float*)d_in[7];
  const float* Wqkv   = (const float*)d_in[8];
  const float* bqkv   = (const float*)d_in[9];
  const float* Wo     = (const float*)d_in[10];
  const float* bo     = (const float*)d_in[11];
  const float* ln1s   = (const float*)d_in[12];
  const float* ln1b   = (const float*)d_in[13];
  const float* W1     = (const float*)d_in[14];
  const float* b1     = (const float*)d_in[15];
  const float* W2     = (const float*)d_in[16];
  const float* b2     = (const float*)d_in[17];
  const float* ln2s   = (const float*)d_in[18];
  const float* ln2b   = (const float*)d_in[19];
  const float* predw  = (const float*)d_in[20];
  const float* predb  = (const float*)d_in[21];
  float* out = (float*)d_out;

  char* ws = (char*)d_ws;
  size_t off = 0;
  auto alloc = [&](size_t bytes) -> char* {
    char* p = ws + off;
    off += (bytes + 255) & ~(size_t)255;
    return p;
  };
  float*  m_f   = (float*)alloc(1024 * 4);
  float*  x_f   = (float*)alloc(ROWE * 4);
  float*  qsh_f = (float*)alloc(ROWE * 4);
  float*  h_f   = (float*)alloc(ROWE * 4);
  float*  g_f   = (float*)alloc(ROWE * 4);
  __bf16* x_h   = (__bf16*)alloc(ROWE * 2);
  __bf16* qsh_h = (__bf16*)alloc(ROWE * 2);
  __bf16* h_h   = (__bf16*)alloc(ROWE * 2);
  __bf16* Q_h   = (__bf16*)alloc(ROWE * 2);
  __bf16* K_h   = (__bf16*)alloc(ROWE * 2);
  __bf16* V_h   = (__bf16*)alloc(ROWE * 2);
  __bf16* O_h   = (__bf16*)alloc(ROWE * 2);
  __bf16* T_h   = (__bf16*)alloc(ROWE * 2);
  __bf16* Wqkv_h = (__bf16*)alloc((size_t)2 * 1536 * 512 * 2);
  __bf16* Wo_h   = (__bf16*)alloc((size_t)2 * 512 * 512 * 2);
  __bf16* W1_h   = (__bf16*)alloc((size_t)2 * 512 * 512 * 2);
  __bf16* W2_h   = (__bf16*)alloc((size_t)2 * 512 * 512 * 2);

  cvt_bf16_kernel<<<512, 256, 0, stream>>>(Wqkv, Wqkv_h, 2 * 1536 * 512);
  cvt_bf16_kernel<<<512, 256, 0, stream>>>(Wo,   Wo_h,   2 * 512 * 512);
  cvt_bf16_kernel<<<512, 256, 0, stream>>>(W1,   W1_h,   2 * 512 * 512);
  cvt_bf16_kernel<<<512, 256, 0, stream>>>(W2,   W2_h,   2 * 512 * 512);
  mask_kernel<<<1, 1024, 0, stream>>>(alphas, gumbel, m_f, out + 8192);
  embed_kernel<<<4096, 256, 0, stream>>>(q, r, qry, inter, ex, pos, x_f, x_h, qsh_f, qsh_h);

  dim3 gg(64, 8);           // M/128 x N/64 for N = 512
  for (int layer = 0; layer < 2; ++layer) {
    const __bf16* Wl = Wqkv_h + (size_t)layer * 1536 * 512;
    const float*  bl = bqkv + (size_t)layer * 1536;
    gemm_kernel<<<gg, 256, 0, stream>>>(qsh_h, Wl,              bl,        nullptr, Q_h, 0);
    gemm_kernel<<<gg, 256, 0, stream>>>(x_h,   Wl + 512 * 512,  bl + 512,  nullptr, K_h, 0);
    gemm_kernel<<<gg, 256, 0, stream>>>(x_h,   Wl + 1024 * 512, bl + 1024, nullptr, V_h, 0);
    attn_kernel<<<dim3(64, 16), 128, 0, stream>>>(Q_h, K_h, V_h, m_f, O_h);
    gemm_kernel<<<gg, 256, 0, stream>>>(O_h, Wo_h + (size_t)layer * 262144, bo + layer * 512,
                                        g_f, nullptr, 0);
    ln_kernel<<<1024, 256, 0, stream>>>(qsh_f, g_f, ln1s + layer * 512, ln1b + layer * 512,
                                        h_f, h_h);
    gemm_kernel<<<gg, 256, 0, stream>>>(h_h, W1_h + (size_t)layer * 262144, b1 + layer * 512,
                                        nullptr, T_h, 1);
    gemm_kernel<<<gg, 256, 0, stream>>>(T_h, W2_h + (size_t)layer * 262144, b2 + layer * 512,
                                        g_f, nullptr, 0);
    ln_kernel<<<1024, 256, 0, stream>>>(h_f, g_f, ln2s + layer * 512, ln2b + layer * 512,
                                        x_f, x_h);
  }
  pred_kernel<<<1024, 256, 0, stream>>>(x_f, predw, predb, out);
}
